// SABlock_39281770889877
// MI455X (gfx1250) — compile-verified, matrix operand order checked by reading, not run on hardware
//
#include <hip/hip_runtime.h>
#include <hip/hip_bf16.h>

typedef unsigned short u16;
typedef unsigned int   u32;
typedef __attribute__((ext_vector_type(16))) __bf16 v16bf;
typedef __attribute__((ext_vector_type(8)))  float  v8f;

// ---------- helpers ----------

__device__ __forceinline__ u16 f2bf_bits(float x) {
    u32 u = __builtin_bit_cast(u32, x);
    u32 r = u + 0x7FFFu + ((u >> 16) & 1u);   // round-to-nearest-even
    return (u16)(r >> 16);
}
__device__ __forceinline__ __bf16 bits2bf(u16 b) {
    return __builtin_bit_cast(__bf16, b);
}

// Load a 16x32 bf16 frag from a per-lane pointer that already includes
// (row + lane-row)*ld + k0 + 8*(lane>=16).  Element e -> k = (e&7) + 16*(e>>3),
// loaded as 8 aligned dword pairs at offsets {0,2,4,6,16,18,20,22}.
__device__ __forceinline__ v16bf load_frag_ptr(const u16* p) {
    v16bf f;
#pragma unroll
    for (int i = 0; i < 8; ++i) {
        int kk = (i < 4) ? (2 * i) : (16 + 2 * (i - 4));
        u32 d = *(const u32*)(p + kk);
        f[2 * i]     = bits2bf((u16)(d & 0xFFFFu));
        f[2 * i + 1] = bits2bf((u16)(d >> 16));
    }
    return f;
}

// Convenience: frag from row-major [rows, ld] bf16 matrix at (row, k0).
// ISA layout: lanes 0-15 hold row r=lane, K {0..7,16..23}; lanes 16-31 hold
// row r=lane-16, K {8..15,24..31}.
__device__ __forceinline__ v16bf load_frag_rm(const u16* base, int ld, int row,
                                              int k0, int lane) {
    int r = lane & 15;
    int g = (lane >> 4) & 1;
    return load_frag_ptr(base + (size_t)(row + r) * ld + k0 + 8 * g);
}

// B-frag for a K=32 x N matrix stored row-major with leading dim ld
// (element (k,n) = base[k*ld + col]); used for the V operand of P*V.
__device__ __forceinline__ v16bf load_fragB_kmajor(const u16* base, int ld,
                                                   int col, int lane) {
    int g = (lane >> 4) & 1;
    v16bf f;
#pragma unroll
    for (int e = 0; e < 16; ++e) {
        int k = (e & 7) + 8 * g + ((e >> 3) << 4);
        f[e] = bits2bf(base[(size_t)k * ld + col]);
    }
    return f;
}

// ---------- kernels ----------

__global__ void cvt_f32_bf16(const float* __restrict__ in, u16* __restrict__ out, int n) {
    int i = blockIdx.x * blockDim.x + threadIdx.x;
    if (i < n) out[i] = f2bf_bits(in[i]);
}

// C[M,N] = A[M,K] * B[N,K]^T.  A,B bf16 row-major.
// If Cf != nullptr: fp32 output with optional bias. Else bf16 output to Cb.
// 64x64 output tile per wave32 (4x4 grid of 16x16 WMMA tiles): 16 WMMAs per
// 8 operand-frag loads per K-step. Base pointers are loop-invariant (k0
// indexing, no pointer PHIs) so loads lower to global_load_b128, not flat.
// launch_bounds(...,1) -> full VGPR budget, no spills. M,N multiples of 64.
__global__ __launch_bounds__(256, 1)
void gemm_bf16_wmma(const u16* __restrict__ A, const u16* __restrict__ B,
                    int M, int N, int K,
                    u16* __restrict__ Cb, float* __restrict__ Cf,
                    const float* __restrict__ bias) {
    int wavesPerBlock = blockDim.x >> 5;
    int wid  = blockIdx.x * wavesPerBlock + (threadIdx.x >> 5);
    int lane = threadIdx.x & 31;
    int tilesN = N >> 6;
    int nTiles = (M >> 6) * tilesN;
    int stride = gridDim.x * wavesPerBlock;
    int r16 = lane & 15;
    int g   = (lane >> 4) & 1;

    for (int t = wid; t < nTiles; t += stride) {
        int tm = (t / tilesN) << 6, tn = (t % tilesN) << 6;
        // loop-invariant per-lane row base pointers
        const u16* pa0 = A + (size_t)(tm + 0 * 16 + r16) * K + 8 * g;
        const u16* pa1 = A + (size_t)(tm + 1 * 16 + r16) * K + 8 * g;
        const u16* pa2 = A + (size_t)(tm + 2 * 16 + r16) * K + 8 * g;
        const u16* pa3 = A + (size_t)(tm + 3 * 16 + r16) * K + 8 * g;
        const u16* pb0 = B + (size_t)(tn + 0 * 16 + r16) * K + 8 * g;
        const u16* pb1 = B + (size_t)(tn + 1 * 16 + r16) * K + 8 * g;
        const u16* pb2 = B + (size_t)(tn + 2 * 16 + r16) * K + 8 * g;
        const u16* pb3 = B + (size_t)(tn + 3 * 16 + r16) * K + 8 * g;

        v8f acc[4][4] = {};
        for (int k0 = 0; k0 < K; k0 += 32) {
            // speculative prefetch of next K tile (global_prefetch_b8);
            // harmless one-past-the-end on the final iteration.
            __builtin_prefetch(pa0 + k0 + 32, 0, 1);
            __builtin_prefetch(pb0 + k0 + 32, 0, 1);
            v16bf a[4];
            a[0] = load_frag_ptr(pa0 + k0);
            a[1] = load_frag_ptr(pa1 + k0);
            a[2] = load_frag_ptr(pa2 + k0);
            a[3] = load_frag_ptr(pa3 + k0);
            {
                v16bf b = load_frag_ptr(pb0 + k0);
#pragma unroll
                for (int i = 0; i < 4; ++i)
                    acc[i][0] = __builtin_amdgcn_wmma_f32_16x16x32_bf16(
                        false, a[i], false, b, (short)0, acc[i][0], false, false);
            }
            {
                v16bf b = load_frag_ptr(pb1 + k0);
#pragma unroll
                for (int i = 0; i < 4; ++i)
                    acc[i][1] = __builtin_amdgcn_wmma_f32_16x16x32_bf16(
                        false, a[i], false, b, (short)0, acc[i][1], false, false);
            }
            {
                v16bf b = load_frag_ptr(pb2 + k0);
#pragma unroll
                for (int i = 0; i < 4; ++i)
                    acc[i][2] = __builtin_amdgcn_wmma_f32_16x16x32_bf16(
                        false, a[i], false, b, (short)0, acc[i][2], false, false);
            }
            {
                v16bf b = load_frag_ptr(pb3 + k0);
#pragma unroll
                for (int i = 0; i < 4; ++i)
                    acc[i][3] = __builtin_amdgcn_wmma_f32_16x16x32_bf16(
                        false, a[i], false, b, (short)0, acc[i][3], false, false);
            }
        }
#pragma unroll
        for (int i = 0; i < 4; ++i) {
#pragma unroll
            for (int r = 0; r < 8; ++r) {
                int row = tm + i * 16 + r + 8 * g;
#pragma unroll
                for (int j = 0; j < 4; ++j) {
                    int col = tn + j * 16 + r16;
                    float v = acc[i][j][r];
                    if (Cf) Cf[(size_t)row * N + col] = v + (bias ? bias[col] : 0.0f);
                    else    Cb[(size_t)row * N + col] = f2bf_bits(v);
                }
            }
        }
    }
}

// Flash attention, causal. qkv: bf16 [B*T, 3C], q|k|v each C wide, head h at
// column h*dh. One wave per (b, h, 32-row query block); K/V frags reused
// across two 16-row Q subtiles. P staged in LDS as bf16. y: bf16 [B*T, C].
#define ATT_C  1024
#define ATT_H  16
#define ATT_DH 64
__global__ __launch_bounds__(256, 1)
void attn_flash_wmma(const u16* __restrict__ qkv, u16* __restrict__ y,
                     int B, int T) {
    const int C3 = 3 * ATT_C;
    int wib  = threadIdx.x >> 5;
    int lane = threadIdx.x & 31;
    int qbPer = T >> 5;                       // 32-row query blocks
    int w = blockIdx.x * (blockDim.x >> 5) + wib;
    if (w >= B * ATT_H * qbPer) return;
    int b     = w / (ATT_H * qbPer);
    int h     = (w / qbPer) % ATT_H;
    int qbase = (w % qbPer) << 5;

    __shared__ u16 pLds[8][2][16 * 32];       // per-wave 2KB bf16 P staging
    u16* P0 = pLds[wib][0];
    u16* P1 = pLds[wib][1];

    const u16* qkvB = qkv + (size_t)b * T * C3;
    const u16* Qb = qkvB + h * ATT_DH;
    const u16* Kb = qkvB + ATT_C + h * ATT_DH;
    const u16* Vb = qkvB + 2 * ATT_C + h * ATT_DH;
    const float scale = 0.125f;               // 1/sqrt(64)

    int g = (lane >> 4) & 1, c = lane & 15;

    v16bf qA[2][2];
#pragma unroll
    for (int u = 0; u < 2; ++u) {
        qA[u][0] = load_frag_rm(Qb, C3, qbase + u * 16, 0,  lane);
        qA[u][1] = load_frag_rm(Qb, C3, qbase + u * 16, 32, lane);
    }

    v8f acc[2][4] = {};
    float m[2][8], l[2][8];
#pragma unroll
    for (int u = 0; u < 2; ++u)
#pragma unroll
        for (int r = 0; r < 8; ++r) { m[u][r] = -INFINITY; l[u][r] = 0.0f; }

    for (int jb = 0; jb <= qbase + 31; jb += 32) {
        // S = Q * K^T: 2 Q-subtiles x 2 key-subtiles; K frags reused across u
        v8f s[2][2];
#pragma unroll
        for (int t = 0; t < 2; ++t) {
            v16bf kb0 = load_frag_rm(Kb, C3, jb + t * 16, 0,  lane);
            v16bf kb1 = load_frag_rm(Kb, C3, jb + t * 16, 32, lane);
#pragma unroll
            for (int u = 0; u < 2; ++u) {
                v8f z = {};
                z = __builtin_amdgcn_wmma_f32_16x16x32_bf16(
                    false, qA[u][0], false, kb0, (short)0, z, false, false);
                z = __builtin_amdgcn_wmma_f32_16x16x32_bf16(
                    false, qA[u][1], false, kb1, (short)0, z, false, false);
                s[u][t] = z;
            }
        }
        // online softmax per row (rows live in 16-lane halves)
#pragma unroll
        for (int u = 0; u < 2; ++u) {
            u16* P = (u == 0) ? P0 : P1;
#pragma unroll
            for (int r = 0; r < 8; ++r) {
                int row = qbase + u * 16 + r + 8 * g;
                float s0 = s[u][0][r] * scale;
                float s1 = s[u][1][r] * scale;
                if (jb + c > row)      s0 = -INFINITY;   // causal mask
                if (jb + 16 + c > row) s1 = -INFINITY;
                float mx = fmaxf(s0, s1);
#pragma unroll
                for (int d = 8; d >= 1; d >>= 1) mx = fmaxf(mx, __shfl_xor(mx, d, 32));
                float mnew  = fmaxf(m[u][r], mx);
                float alpha = __expf(m[u][r] - mnew);
                float p0 = __expf(s0 - mnew);
                float p1 = __expf(s1 - mnew);
                float rs = p0 + p1;
#pragma unroll
                for (int d = 8; d >= 1; d >>= 1) rs += __shfl_xor(rs, d, 32);
                l[u][r] = l[u][r] * alpha + rs;
                m[u][r] = mnew;
                P[(r + 8 * g) * 32 + c]      = f2bf_bits(p0);
                P[(r + 8 * g) * 32 + 16 + c] = f2bf_bits(p1);
#pragma unroll
                for (int nt = 0; nt < 4; ++nt) acc[u][nt][r] *= alpha;
            }
        }
        __builtin_amdgcn_wave_barrier();
        asm volatile("" ::: "memory");        // DS pipe is in-order per wave
        // O += P * V  (K = 32 keys, N = 64 = 4 tiles); V frags reused across u
        v16bf pA[2];
        pA[0] = load_frag_rm(P0, 32, 0, 0, lane);   // ds_load_b32 x8
        pA[1] = load_frag_rm(P1, 32, 0, 0, lane);
#pragma unroll
        for (int nt = 0; nt < 4; ++nt) {
            v16bf vb = load_fragB_kmajor(Vb + (size_t)jb * C3, C3,
                                         nt * 16 + c, lane);
#pragma unroll
            for (int u = 0; u < 2; ++u)
                acc[u][nt] = __builtin_amdgcn_wmma_f32_16x16x32_bf16(
                    false, pA[u], false, vb, (short)0, acc[u][nt], false, false);
        }
        __builtin_amdgcn_wave_barrier();
        asm volatile("" ::: "memory");
    }

    // y[b, qbase + u*16 + row, h*dh + col] = acc / l
    u16* yB = y + ((size_t)b * T + qbase) * ATT_C + h * ATT_DH;
#pragma unroll
    for (int u = 0; u < 2; ++u) {
#pragma unroll
        for (int r = 0; r < 8; ++r) {
            float inv = 1.0f / l[u][r];
#pragma unroll
            for (int nt = 0; nt < 4; ++nt)
                yB[(size_t)(u * 16 + r + 8 * g) * ATT_C + nt * 16 + c] =
                    f2bf_bits(acc[u][nt][r] * inv);
        }
    }
}

// ---------- launch ----------

extern "C" void kernel_launch(void* const* d_in, const int* in_sizes, int n_in,
                              void* d_out, int out_size, void* d_ws, size_t ws_size,
                              hipStream_t stream) {
    const float* x     = (const float*)d_in[0];   // [B,T,C]
    const float* w_qkv = (const float*)d_in[1];   // [3C,C]
    const float* w_out = (const float*)d_in[2];   // [C,C]
    const float* b_out = (const float*)d_in[3];   // [C]
    float* out = (float*)d_out;

    const int B = 2, T = 2048, C = 1024;
    const int M = B * T;                          // 4096 tokens

    // workspace layout (bf16 = u16)
    u16* xb    = (u16*)d_ws;                      // M*C
    u16* wqkvb = xb    + (size_t)M * C;           // 3C*C
    u16* woutb = wqkvb + (size_t)3 * C * C;       // C*C
    u16* qkvb  = woutb + (size_t)C * C;           // M*3C
    u16* yb    = qkvb  + (size_t)M * 3 * C;       // M*C

    // 1) fp32 -> bf16 conversions
    {
        int n = M * C;
        cvt_f32_bf16<<<(n + 255) / 256, 256, 0, stream>>>(x, xb, n);
        n = 3 * C * C;
        cvt_f32_bf16<<<(n + 255) / 256, 256, 0, stream>>>(w_qkv, wqkvb, n);
        n = C * C;
        cvt_f32_bf16<<<(n + 255) / 256, 256, 0, stream>>>(w_out, woutb, n);
    }
    // 2) qkv = x @ w_qkv^T  -> bf16 [M, 3C]
    {
        int tiles  = (M / 64) * (3 * C / 64);     // 64x48 = 3072 waves
        int blocks = (tiles + 7) / 8;             // 8 waves/block
        gemm_bf16_wmma<<<blocks, 256, 0, stream>>>(xb, wqkvb, M, 3 * C, C,
                                                   qkvb, nullptr, nullptr);
    }
    // 3) flash attention -> bf16 y [M, C]
    {
        int waves  = B * 16 * (T / 32);           // 2048
        int blocks = (waves + 7) / 8;
        attn_flash_wmma<<<blocks, 256, 0, stream>>>(qkvb, yb, B, T);
    }
    // 4) out = y @ w_out^T + b_out -> fp32
    {
        int tiles  = (M / 64) * (C / 64);         // 64x16 = 1024 waves
        int blocks = (tiles + 7) / 8;
        gemm_bf16_wmma<<<blocks, 256, 0, stream>>>(yb, woutb, M, C, C,
                                                   nullptr, out, b_out);
    }
}
